// AdaptiveBlockSelfAttention_72842645340578
// MI455X (gfx1250) — matrix operand order repaired and checked by reading, not compiled, not run on hardware
//
#include <hip/hip_runtime.h>

// ---------------------------------------------------------------------------
// Block-local self-attention, fully WMMA-based for gfx1250 (MI455X).
//   K0: convert Wq/Wk/Wv/Wo f32 -> f16 once into workspace tail (512 KB)
//   K1: per (block,head,batch) WG: x->LDS(f16, fragment-swizzled), QKV GEMM,
//       attention, O->ws(f16).  B operands live in LDS in WMMA-fragment-native
//       slot layout [tile][lane][16 halfs] so every B fragment is one 32-byte
//       contiguous, bank-conflict-free LDS load.
//   K2: per pixel-tile WG: Wo GEMM + residual + LayerNorm -> out(f32)
// All matmuls: v_wmma_f32_16x16x32_f16 (f32 accumulate).
// ---------------------------------------------------------------------------

typedef __attribute__((ext_vector_type(16))) _Float16 v16h;
typedef __attribute__((ext_vector_type(8)))  _Float16 v8h;
typedef __attribute__((ext_vector_type(4)))  _Float16 v4h;
typedef __attribute__((ext_vector_type(8)))  float    v8f;
typedef __attribute__((ext_vector_type(4)))  float    v4f;

#define PITCH_S  264   // sS strips [16 i][264]  (row-major, A-frag reads)
#define PITCH_QV 40    // sQ [256 px][40]        (row-major, A-frag reads)

#define WS_O_HALFS (2L * 36864L * 256L)   // O scratch: [b*36864+pix][256c] f16

__device__ __forceinline__ v8f wmma_f16(v16h a, v16h b, v8f c) {
  return __builtin_amdgcn_wmma_f32_16x16x32_f16(false, a, false, b,
                                                (short)0, c, false, false);
}

// A fragment (16x32 f16): lane row m, K = {16*half..+16} (contiguous).
// Works for LDS or global f16 (two contiguous 16B loads per lane).
__device__ __forceinline__ v16h load_A_frag(const _Float16* base, int pitch,
                                            int m, int half) {
  const _Float16* r = base + (long)m * pitch;
  v8h lo = *(const v8h*)(r + 16 * half);
  v8h hi = *(const v8h*)(r + 16 * half + 8);
  v16h a;
#pragma unroll
  for (int e = 0; e < 8; ++e) { a[e] = lo[e]; a[8 + e] = hi[e]; }
  return a;
}

// B fragment from fragment-swizzled LDS: one 32B contiguous load per lane.
// slot holds the 16 K-values (e=0..15) for this lane's (N=n, Khalf=half).
__device__ __forceinline__ v16h lds_load_B_sw(const _Float16* slot) {
  return *(const v16h*)slot;
}

// Store C/D tile transposed: row = N (lane col n), col = M = r + 8*half.
// Per lane: 8 contiguous halfs -> one 16B ds_store.
__device__ __forceinline__ void store_D_T(_Float16* base, int pitch, int n,
                                          int half, v8f d) {
  v8h h;
#pragma unroll
  for (int r = 0; r < 8; ++r) h[r] = (_Float16)d[r];
  *(v8h*)(base + n * pitch + 8 * half) = h;
}

// Store C/D tile row-major: row = M = r + 8*half, col = n (8 scalar stores).
__device__ __forceinline__ void store_D_rm(_Float16* base, int pitch, int n,
                                           int half, v8f d) {
#pragma unroll
  for (int r = 0; r < 8; ++r)
    base[(r + 8 * half) * pitch + n] = (_Float16)d[r];
}

// ---------------------------------------------------------------------------
// Kernel 0: one-time f32 -> f16 conversion of the four 256x256 weights.
// ---------------------------------------------------------------------------
__global__ __launch_bounds__(256)
void k_cvt_w(const float* __restrict__ Wq, const float* __restrict__ Wk,
             const float* __restrict__ Wv, const float* __restrict__ Wo,
             _Float16* __restrict__ dst) {
  const int i = (blockIdx.x * 256 + threadIdx.x) * 4;   // 0..65532
  const float* srcs[4] = {Wq, Wk, Wv, Wo};
#pragma unroll
  for (int w = 0; w < 4; ++w) {
    v4f f = *(const v4f*)(srcs[w] + i);
    v4h h;
#pragma unroll
    for (int e = 0; e < 4; ++e) h[e] = (_Float16)f[e];
    *(v4h*)(dst + (long)w * 65536 + i) = h;
  }
}

// ---------------------------------------------------------------------------
// Kernel 1: QKV projection + block-local attention for one (block, head, b).
// grid = (144 blocks, 8 heads, 2 batch), 256 threads (8 wave32).
// LDS (halfs): sXs 65536 | sQ 10240 | sVs 8192 | sKs 8192 | sS 33792
//            = 125952 halfs = 246 KB dynamic.
// Swizzled slot index convention (16 halfs per slot, 32 slots per 32x16 tile):
//   sXs[((kt*16+nt)*32 + cH*16 + pn)*16 + e] = x[c=kt*32+cH*16+e][px=nt*16+pn]
//   sKs[((nt*32 + dH*16 + jn)*16 + e]        = K[d=dH*16+e][j=nt*16+jn]
//   sVs[((jt*32 + d)*16 + jn]                = V[j=jt*16+jn][d]
// ---------------------------------------------------------------------------
__global__ __launch_bounds__(256)
void k_qkv_attn(const float* __restrict__ x, const _Float16* __restrict__ w16,
                _Float16* __restrict__ ws) {
  extern __shared__ char smem[];
  _Float16* sXs = (_Float16*)smem;            // 65536 halfs
  _Float16* sQ  = sXs + 65536;                // [256 px][PITCH_QV]
  _Float16* sVs = sQ + 256 * PITCH_QV;        // 8192 halfs
  _Float16* sKs = sVs + 8192;                 // 8192 halfs
  _Float16* sS  = sKs + 8192;                 // [8 waves][16 i][PITCH_S]

  const int tid = threadIdx.x;
  const int lane = tid & 31, wv = tid >> 5;
  const int n = lane & 15, half = lane >> 4;

  const int blk  = blockIdx.x;      // 0..143
  const int head = blockIdx.y;      // 0..7
  const int b    = blockIdx.z;      // 0..1
  const int bh = blk / 12, bw = blk % 12;
  const long pixBase = (long)(bh * 16) * 192 + bw * 16;
  const float* xb = x + (long)b * 256 * 36864;

  // ---- 1) stage x tile into fragment-swizzled f16 LDS ----------------------
  // thread = channel c; pixel row py is also the pixel-tile index (px>>4==py).
  {
    const int c = tid;
    const int kt = c >> 5, cH = (c >> 4) & 1, e = c & 15;
    const float* src = xb + (long)c * 36864 + pixBase;
    _Float16* dstb = sXs + (kt * 512 + cH * 16) * 16 + e;
#pragma unroll 4
    for (int py = 0; py < 16; ++py) {
      const float* rp = src + py * 192;
      v4f f0 = *(const v4f*)(rp);
      v4f f1 = *(const v4f*)(rp + 4);
      v4f f2 = *(const v4f*)(rp + 8);
      v4f f3 = *(const v4f*)(rp + 12);
      _Float16* drow = dstb + (py * 32) * 16;
#pragma unroll
      for (int q = 0; q < 4; ++q) {
        drow[(q * 4 + 0) * 16] = (_Float16)((q==0?f0:q==1?f1:q==2?f2:f3)[0]);
        drow[(q * 4 + 1) * 16] = (_Float16)((q==0?f0:q==1?f1:q==2?f2:f3)[1]);
        drow[(q * 4 + 2) * 16] = (_Float16)((q==0?f0:q==1?f1:q==2?f2:f3)[2]);
        drow[(q * 4 + 3) * 16] = (_Float16)((q==0?f0:q==1?f1:q==2?f2:f3)[3]);
      }
    }
  }
  __syncthreads();

  // ---- 2) Q/K/V for this head: [32 o x 256 c] * [256 c x 256 px] -----------
  // 96 output tiles (3 proj x 2 Mtiles x 16 Ntiles) across 8 waves.
  for (int task = wv; task < 96; task += 8) {
    const int p  = task >> 5;           // 0=Q 1=K 2=V
    const int mt = (task >> 4) & 1;     // o tile within head
    const int nt = task & 15;           // pixel tile
    const _Float16* W = w16 + (long)p * 65536;   // Wq/Wk/Wv f16
    const int oBase = head * 32 + mt * 16;
    v8f acc = {0.f, 0.f, 0.f, 0.f, 0.f, 0.f, 0.f, 0.f};
#pragma unroll
    for (int kt = 0; kt < 8; ++kt) {
      v16h a  = load_A_frag(W + (long)oBase * 256 + kt * 32, 256, n, half);
      v16h bB = lds_load_B_sw(sXs + ((kt * 16 + nt) * 32 + half * 16 + n) * 16);
      acc = wmma_f16(a, bB, acc);
    }
    if (p == 0) {
      store_D_T(sQ + (nt * 16) * PITCH_QV + mt * 16, PITCH_QV, n, half, acc);
    } else if (p == 1) {
      // K: D tile maps exactly onto swizzled slots -> one 16B store per lane
      v8h h;
#pragma unroll
      for (int r = 0; r < 8; ++r) h[r] = (_Float16)acc[r];
      *(v8h*)(sKs + (nt * 32 + mt * 16 + n) * 16 + 8 * half) = h;
    } else {
      // V: scatter into swizzled slots (8 scalar stores per lane)
#pragma unroll
      for (int r = 0; r < 8; ++r)
        sVs[(nt * 32 + mt * 16 + 8 * half + r) * 16 + n] = (_Float16)acc[r];
    }
  }
  __syncthreads();

  // ---- 3) attention; each wave owns 2 i-tiles (32 rows of S) ---------------
  _Float16* sSw = sS + wv * 16 * PITCH_S;
  const float scale = 0.17677669529663687f;  // 1/sqrt(32)

  for (int rep = 0; rep < 2; ++rep) {
    const int iBase = (wv * 2 + rep) * 16;

    // S tile row strip: one WMMA per j-tile (K = d = 32 exactly)
    v16h aQ = load_A_frag(sQ + iBase * PITCH_QV, PITCH_QV, n, half);
    v8f s[16];
#pragma unroll
    for (int jt = 0; jt < 16; ++jt) {
      s[jt] = (v8f){0.f, 0.f, 0.f, 0.f, 0.f, 0.f, 0.f, 0.f};
      v16h bK = lds_load_B_sw(sKs + (jt * 32 + half * 16 + n) * 16);
      s[jt] = wmma_f16(aQ, bK, s[jt]);
    }

    // softmax over j (16 tiles in-lane, then 16 lanes of the half-wave)
    float mx[8], sm[8];
#pragma unroll
    for (int r = 0; r < 8; ++r) mx[r] = -3.0e38f;
#pragma unroll
    for (int jt = 0; jt < 16; ++jt)
#pragma unroll
      for (int r = 0; r < 8; ++r) {
        s[jt][r] *= scale;
        mx[r] = fmaxf(mx[r], s[jt][r]);
      }
#pragma unroll
    for (int r = 0; r < 8; ++r) {
#pragma unroll
      for (int d = 1; d < 16; d <<= 1)
        mx[r] = fmaxf(mx[r], __shfl_xor(mx[r], d, 32));
      sm[r] = 0.f;
    }
#pragma unroll
    for (int jt = 0; jt < 16; ++jt)
#pragma unroll
      for (int r = 0; r < 8; ++r) {
        float e = __expf(s[jt][r] - mx[r]);
        s[jt][r] = e;
        sm[r] += e;
      }
#pragma unroll
    for (int r = 0; r < 8; ++r) {
#pragma unroll
      for (int d = 1; d < 16; d <<= 1)
        sm[r] += __shfl_xor(sm[r], d, 32);
      sm[r] = 1.f / sm[r];
    }
    // normalized attn -> LDS strip (C/D -> A-layout conversion)
#pragma unroll
    for (int jt = 0; jt < 16; ++jt) {
      v8f pnorm;
#pragma unroll
      for (int r = 0; r < 8; ++r) pnorm[r] = s[jt][r] * sm[r];
      store_D_rm(sSw + jt * 16, PITCH_S, n, half, pnorm);
    }

    // O = attn * V : [16 i x 256 j] * [256 j x 32 d]
#pragma unroll
    for (int dt = 0; dt < 2; ++dt) {
      v8f o = {0.f, 0.f, 0.f, 0.f, 0.f, 0.f, 0.f, 0.f};
#pragma unroll
      for (int ks = 0; ks < 8; ++ks) {
        v16h aS = load_A_frag(sSw + ks * 32, PITCH_S, n, half);
        v16h bV = lds_load_B_sw(sVs + ((ks * 2 + half) * 32 + dt * 16 + n) * 16);
        o = wmma_f16(aS, bV, o);
      }
      // write O to workspace f16, layout [b*36864 + pix][256 c]
#pragma unroll
      for (int r = 0; r < 8; ++r) {
        const int i = iBase + r + 8 * half;            // local pixel
        const long pix = pixBase + (long)(i >> 4) * 192 + (i & 15);
        ws[((long)b * 36864 + pix) * 256 + head * 32 + dt * 16 + n] =
            (_Float16)o[r];
      }
    }
  }
}

// ---------------------------------------------------------------------------
// Kernel 2: Y = Wo*O + x, LayerNorm over channels, out f32.
// grid = 576 WGs of 256 threads; each WG does 128 consecutive pixels.
// ---------------------------------------------------------------------------
__global__ __launch_bounds__(256)
void k_proj_ln(const _Float16* __restrict__ ws, const _Float16* __restrict__ wo16,
               const float* __restrict__ x, const float* __restrict__ gamma,
               const float* __restrict__ beta, float* __restrict__ out) {
  __shared__ float sSum[8][128];
  __shared__ float sSq[8][128];
  __shared__ float sMu[128];
  __shared__ float sRs[128];

  const int tid = threadIdx.x;
  const int lane = tid & 31, wv = tid >> 5;
  const int n = lane & 15, half = lane >> 4;

  const long pixBase = (long)blockIdx.x * 128;
  const int  b    = (int)(pixBase / 36864);
  const long pixb = pixBase % 36864;
  const float* xb = x   + (long)b * 256 * 36864 + pixb;
  float*       ob = out + (long)b * 256 * 36864 + pixb;
  const int oBase = wv * 32;  // wave owns 32 output channels

  // pull the O tile for this WG toward the caches (global_prefetch_b8)
  __builtin_prefetch(ws + (pixBase + tid) * 256, 0, 0);

  v8f acc[2][8];
#pragma unroll
  for (int ot = 0; ot < 2; ++ot)
#pragma unroll
    for (int pt = 0; pt < 8; ++pt)
      acc[ot][pt] = (v8f){0.f, 0.f, 0.f, 0.f, 0.f, 0.f, 0.f, 0.f};

#pragma unroll 2
  for (int kt = 0; kt < 8; ++kt) {
    v16h a0 = load_A_frag(wo16 + (long)oBase * 256 + kt * 32, 256, n, half);
    v16h a1 = load_A_frag(wo16 + (long)(oBase + 16) * 256 + kt * 32, 256, n, half);
#pragma unroll
    for (int pt = 0; pt < 8; ++pt) {
      // B fragment direct from ws: 32 contiguous-per-lane bytes (L0/L2 hit)
      v16h bB = *(const v16h*)(ws + (pixBase + pt * 16 + n) * 256 +
                               kt * 32 + 16 * half);
      acc[0][pt] = wmma_f16(a0, bB, acc[0][pt]);
      acc[1][pt] = wmma_f16(a1, bB, acc[1][pt]);
    }
  }

  // residual add + per-pixel partial stats over this wave's 32 channels
  float psum[8], psq[8];
#pragma unroll
  for (int pt = 0; pt < 8; ++pt) { psum[pt] = 0.f; psq[pt] = 0.f; }
#pragma unroll
  for (int ot = 0; ot < 2; ++ot) {
    const int oT = oBase + ot * 16;
#pragma unroll
    for (int r = 0; r < 8; ++r) {
      const int o = oT + r + 8 * half;
      const float* xrow = xb + (long)o * 36864;
#pragma unroll
      for (int pt = 0; pt < 8; ++pt) {
        float v = acc[ot][pt][r] + xrow[pt * 16 + n];
        acc[ot][pt][r] = v;
        psum[pt] += v;
        psq[pt]  += v * v;
      }
    }
  }
#pragma unroll
  for (int pt = 0; pt < 8; ++pt) {
    psum[pt] += __shfl_xor(psum[pt], 16, 32);
    psq[pt]  += __shfl_xor(psq[pt], 16, 32);
  }
  if (half == 0) {
#pragma unroll
    for (int pt = 0; pt < 8; ++pt) {
      sSum[wv][pt * 16 + n] = psum[pt];
      sSq[wv][pt * 16 + n]  = psq[pt];
    }
  }
  __syncthreads();
  if (tid < 128) {
    float s = 0.f, q = 0.f;
#pragma unroll
    for (int w = 0; w < 8; ++w) { s += sSum[w][tid]; q += sSq[w][tid]; }
    const float mu = s * (1.f / 256.f);
    const float var = q * (1.f / 256.f) - mu * mu;
    sMu[tid] = mu;
    sRs[tid] = rsqrtf(var + 1e-5f);
  }
  __syncthreads();

  // normalize, gamma/beta, coalesced f32 store
#pragma unroll
  for (int ot = 0; ot < 2; ++ot) {
#pragma unroll
    for (int r = 0; r < 8; ++r) {
      const int o = oBase + ot * 16 + r + 8 * half;
      const float g = gamma[o];
      const float be = beta[o];
      float* orow = ob + (long)o * 36864;
#pragma unroll
      for (int pt = 0; pt < 8; ++pt) {
        const int p = pt * 16 + n;
        orow[p] = (acc[ot][pt][r] - sMu[p]) * sRs[p] * g + be;
      }
    }
  }
}

// ---------------------------------------------------------------------------
extern "C" void kernel_launch(void* const* d_in, const int* in_sizes, int n_in,
                              void* d_out, int out_size, void* d_ws,
                              size_t ws_size, hipStream_t stream) {
  const float* x     = (const float*)d_in[0];
  const float* Wq    = (const float*)d_in[1];
  const float* Wk    = (const float*)d_in[2];
  const float* Wv    = (const float*)d_in[3];
  const float* Wo    = (const float*)d_in[4];
  const float* gamma = (const float*)d_in[5];
  const float* beta  = (const float*)d_in[6];
  float* out = (float*)d_out;

  _Float16* ws  = (_Float16*)d_ws;        // O scratch: 37.75 MB
  _Float16* w16 = ws + WS_O_HALFS;        // f16 weights: 4 x 128 KB
  _Float16* wo16 = w16 + 3L * 65536;

  // K0: one-time weight conversion (64 WGs)
  k_cvt_w<<<dim3(64), 256, 0, stream>>>(Wq, Wk, Wv, Wo, w16);

  const size_t smem1 =
      (size_t)(65536 + 256 * PITCH_QV + 8192 + 8192 + 8 * 16 * PITCH_S) *
      sizeof(_Float16);

  dim3 g1(144, 8, 2);
  k_qkv_attn<<<g1, 256, smem1, stream>>>(x, w16, ws);

  dim3 g2(576, 1, 1);
  k_proj_ln<<<g2, 256, 0, stream>>>(ws, wo16, x, gamma, beta, out);
}
